// FP_SA_20366734918003
// MI455X (gfx1250) — matrix operand order repaired and checked by reading, not compiled, not run on hardware
//
#include <hip/hip_runtime.h>
#include <hip/hip_bf16.h>
#include <stdint.h>

#define BB 4
#define NN 16384
#define SS 16384
#define NH 8

typedef __bf16 bf;
typedef __attribute__((ext_vector_type(16))) __bf16 v16bf;
typedef __attribute__((ext_vector_type(8)))  float  v8f;
typedef __attribute__((__vector_size__(16))) int v4i;

#define AS1 __attribute__((address_space(1)))
#define AS3 __attribute__((address_space(3)))

#if __has_builtin(__builtin_amdgcn_global_load_async_to_lds_b128) && \
    __has_builtin(__builtin_amdgcn_global_store_async_from_lds_b128) && \
    __has_builtin(__builtin_amdgcn_s_wait_asynccnt)
#define HAVE_ASYNC 1
#else
#define HAVE_ASYNC 0
#endif

// 16-byte global -> LDS copy (ASYNCcnt-tracked on gfx1250)
__device__ __forceinline__ void cp_async16(void* lds, const void* g){
#if HAVE_ASYNC
  __builtin_amdgcn_global_load_async_to_lds_b128(
      (AS1 v4i*)(uintptr_t)g, (AS3 v4i*)(uint32_t)(uintptr_t)lds, 0, 0);
#else
  *(float4*)lds = *(const float4*)g;
#endif
}
// 16-byte LDS -> global copy
__device__ __forceinline__ void st_async16(void* g, const void* lds){
#if HAVE_ASYNC
  __builtin_amdgcn_global_store_async_from_lds_b128(
      (AS1 v4i*)(uintptr_t)g, (AS3 v4i*)(uint32_t)(uintptr_t)lds, 0, 0);
#else
  *(float4*)g = *(const float4*)lds;
#endif
}
__device__ __forceinline__ void wait_async(){
#if HAVE_ASYNC
  __builtin_amdgcn_s_wait_asynccnt(0);
#endif
}

__device__ __forceinline__ bf f2bf(float f){
  union { float f; uint32_t u; } v; v.f = f;
  uint32_t r = v.u + 0x7FFFu + ((v.u >> 16) & 1u);
  uint16_t h = (uint16_t)(r >> 16);
  bf out; __builtin_memcpy(&out, &h, 2); return out;
}
__device__ __forceinline__ float bf2f(bf b){
  uint16_t h; __builtin_memcpy(&h, &b, 2);
  union { uint32_t u; float f; } v; v.u = ((uint32_t)h) << 16; return v.f;
}

// ---- WMMA fragment loaders (bf16 16x16x32, wave32; layouts per ISA 7.12.2) ----
__device__ __forceinline__ v16bf load_a(const bf* p, int ld, int lane){
  int r = lane & 15; int kq = (lane >> 4) << 3;
  const bf* q = p + r * ld + kq;
  v16bf a;
#pragma unroll
  for (int i = 0; i < 8; ++i) a[i] = q[i];
#pragma unroll
  for (int i = 0; i < 8; ++i) a[8 + i] = q[16 + i];
  return a;
}
__device__ __forceinline__ v16bf load_b(const bf* p, int ld, int lane){
  int n = lane & 15; int kb = (lane >> 4) << 3;
  const bf* q = p + n * ld + kb;
  v16bf b;
#pragma unroll
  for (int i = 0; i < 8; ++i) b[i] = q[i];
#pragma unroll
  for (int i = 0; i < 8; ++i) b[8 + i] = q[16 + i];
  return b;
}
__device__ __forceinline__ v8f wmma_bf16(v16bf a, v16bf b, v8f c){
  return __builtin_amdgcn_wmma_f32_16x16x32_bf16(false, a, false, b, (short)0, c, false, false);
}

// Column-strip GEMM, K=256: cache the 8 B-fragments in VGPRs, sweep 8 M-tiles.
template<class Epi>
__device__ __forceinline__ void gemm_strip_k256(const bf* Alds, const bf* Wrow,
                                                int nt, int lane, Epi epi){
  v16bf bw[8];
#pragma unroll
  for (int k8 = 0; k8 < 8; ++k8) bw[k8] = load_b(Wrow + nt*16*256 + k8*32, 256, lane);
#pragma unroll
  for (int mt = 0; mt < 8; ++mt){
    v8f acc = {};
#pragma unroll
    for (int k8 = 0; k8 < 8; ++k8)
      acc = wmma_bf16(load_a(Alds + (mt*16)*256 + k8*32, 256, lane), bw[k8], acc);
    epi(mt, acc);
  }
}

// ---------------------------------------------------------------------------
__global__ void k_cvt(const float* __restrict__ s, bf* __restrict__ d, int n){
  int i = blockIdx.x * 256 + threadIdx.x;
  if (i < n) d[i] = f2bf(s[i]);
}

// [B,256,16384] f32 -> [B,16384,256] bf16, LDS-tiled 64x64 (coalesced both ways)
__global__ void k_transpose(const float* __restrict__ s, bf* __restrict__ d){
  __shared__ float tile[64][65];
  int tid = threadIdx.x;
  int nblk = blockIdx.x & 255;
  int cblk = (blockIdx.x >> 8) & 3;
  int b    = blockIdx.x >> 10;
  int n0 = nblk * 64, c0 = cblk * 64;
  int tn = tid & 63, tg = tid >> 6;
#pragma unroll 4
  for (int i = 0; i < 16; ++i){
    int c = tg * 16 + i;
    tile[c][tn] = s[((size_t)b * 256 + c0 + c) * NN + n0 + tn];
  }
  __syncthreads();
#pragma unroll 4
  for (int i = 0; i < 16; ++i){
    int row = tg * 16 + i;
    d[((size_t)b * NN + n0 + row) * 256 + c0 + tn] = f2bf(tile[tn][row]);
  }
}

// f2p = f2 + relu(xyz2@w1^T + b1)@w2^T + b2   (async-staged in/out)
__global__ void k_pos(const float* __restrict__ xyz2,
                      const float* __restrict__ pw1, const float* __restrict__ pb1,
                      const bf* __restrict__ wpos2, const float* __restrict__ pb2,
                      const bf* __restrict__ f2b, bf* __restrict__ f2pb){
  extern __shared__ char smem[];
  float* xyzl = (float*)smem;                  // 2 KB
  bf*    hid  = (bf*)(smem + 2048);            // 128x256 (64 KB)
  bf*    ft   = (bf*)(smem + 67584);           // 128x256 (64 KB)
  bf*    fout = (bf*)(smem + 133120);          // 128x256 (64 KB)
  int tid = threadIdx.x, lane = tid & 31, wave = tid >> 5;
  int b = blockIdx.x >> 7;
  int s0 = (blockIdx.x & 127) * 128;
  size_t gB = ((size_t)b * SS + s0) * 256 * 2;       // byte offset of tile
  for (int u = tid; u < 4096; u += 256)
    cp_async16((char*)ft + u*16, (const char*)f2b + gB + u*16);
  for (int i = tid; i < 384; i += 256) xyzl[i] = xyz2[((size_t)b * SS + s0) * 3 + i];
  __syncthreads();
  {
    int dc = tid;
    float w0 = pw1[dc*3], w1 = pw1[dc*3+1], w2 = pw1[dc*3+2], bb0 = pb1[dc];
    for (int r = 0; r < 128; ++r){
      float v = xyzl[r*3]*w0 + xyzl[r*3+1]*w1 + xyzl[r*3+2]*w2 + bb0;
      hid[r*256 + dc] = f2bf(v > 0.f ? v : 0.f);
    }
  }
  wait_async();
  __syncthreads();
  int rb2 = (lane >> 4) << 3;
  for (int which = 0; which < 2; ++which){
    int nt = wave + which * 8;
    int col = nt*16 + (lane & 15);
    float bias = pb2[col];
    gemm_strip_k256(hid, wpos2, nt, lane, [&](int mt, v8f acc){
      int rbase = mt*16 + rb2;
#pragma unroll
      for (int r = 0; r < 8; ++r){
        int idx = (rbase + r)*256 + col;
        fout[idx] = f2bf(acc[r] + bias + bf2f(ft[idx]));
      }
    });
  }
  __syncthreads();
  for (int u = tid; u < 4096; u += 256)
    st_async16((char*)f2pb + gB + u*16, (const char*)fout + u*16);
  wait_async();
}

// K = elu(f2@wk^T)+1 ; V = f2p@wv^T / S ; KV[h] += K_h^T V_h ; Ksum += sum_s K
__global__ void k_kv(const bf* __restrict__ f2b, const bf* __restrict__ f2pb,
                     const bf* __restrict__ wk, const bf* __restrict__ wv,
                     float* __restrict__ KVg, float* __restrict__ Ksumg){
  extern __shared__ char smem[];
  bf* ft  = (bf*)smem;                 // 128x256
  bf* fpt = (bf*)(smem + 65536);       // 128x256
  bf* Kt  = (bf*)(smem + 131072);      // 256x128  [d][s]
  bf* Vt  = (bf*)(smem + 196608);      // 256x128  [v'][s]
  int tid = threadIdx.x, lane = tid & 31, h = tid >> 5;   // wave == head
  int b = blockIdx.x >> 4;
  int sbase = (blockIdx.x & 15) * 1024;
  v8f kv0 = {}, kv1 = {}, kv2 = {}, kv3 = {};
  float ks = 0.f;
  int rb2 = (lane >> 4) << 3;
  for (int chunk = 0; chunk < 8; ++chunk){
    int s0 = sbase + chunk * 128;
    size_t gB = ((size_t)b * SS + s0) * 256 * 2;
    for (int u = tid; u < 4096; u += 256){
      cp_async16((char*)ft  + u*16, (const char*)f2b  + gB + u*16);
      cp_async16((char*)fpt + u*16, (const char*)f2pb + gB + u*16);
    }
    wait_async();
    __syncthreads();
    for (int which = 0; which < 2; ++which){
      int nt = h + which * 8;
      int col = nt*16 + (lane & 15);
      gemm_strip_k256(ft, wk, nt, lane, [&](int mt, v8f acc){
        int rbase = mt*16 + rb2;
#pragma unroll
        for (int r = 0; r < 8; ++r){
          float kx = acc[r]; kx = kx > 0.f ? kx + 1.f : __expf(kx);
          Kt[col*128 + rbase + r] = f2bf(kx);
        }
      });
      gemm_strip_k256(fpt, wv, nt, lane, [&](int mt, v8f acc){
        int rbase = mt*16 + rb2;
#pragma unroll
        for (int r = 0; r < 8; ++r)
          Vt[col*128 + rbase + r] = f2bf(acc[r] * (1.0f/16384.0f));
      });
    }
    __syncthreads();
    {
      float s = 0.f;
      for (int j = 0; j < 128; ++j) s += bf2f(Kt[tid*128 + j]);
      ks += s;
    }
#pragma unroll
    for (int kk = 0; kk < 128; kk += 32){
      v16bf a0 = load_a(Kt + (h*32     )*128 + kk, 128, lane);
      v16bf a1 = load_a(Kt + (h*32 + 16)*128 + kk, 128, lane);
      v16bf b0 = load_b(Vt + (h*32     )*128 + kk, 128, lane);
      v16bf b1 = load_b(Vt + (h*32 + 16)*128 + kk, 128, lane);
      kv0 = wmma_bf16(a0, b0, kv0);
      kv1 = wmma_bf16(a0, b1, kv1);
      kv2 = wmma_bf16(a1, b0, kv2);
      kv3 = wmma_bf16(a1, b1, kv3);
    }
    __syncthreads();
  }
  int vp = lane & 15;
  float* KVh = KVg + ((size_t)(b * NH + h)) * 32 * 32;
#pragma unroll
  for (int r = 0; r < 8; ++r){
    atomicAdd(&KVh[(rb2 + r     )*32 + vp     ], kv0[r]);
    atomicAdd(&KVh[(rb2 + r     )*32 + vp + 16], kv1[r]);
    atomicAdd(&KVh[(rb2 + r + 16)*32 + vp     ], kv2[r]);
    atomicAdd(&KVh[(rb2 + r + 16)*32 + vp + 16], kv3[r]);
  }
  atomicAdd(&Ksumg[b*256 + tid], ks);
}

// Q proj + linear-attention apply + merge GEMM + LayerNorm1
__global__ void k_attn(const bf* __restrict__ f1b,
                       const bf* __restrict__ wq, const bf* __restrict__ wmg,
                       const float* __restrict__ KVg, const float* __restrict__ Ksumg,
                       const float* __restrict__ g1, const float* __restrict__ b1,
                       bf* __restrict__ ln1b){
  extern __shared__ char smem[];
  bf*    f1t  = (bf*)smem;                 // 64 KB (aliased by msgT)
  bf*    Qt   = (bf*)(smem + 65536);       // 64 KB (aliased by lnT)
  bf*    KVt  = (bf*)(smem + 131072);      // 16 KB, [h][v'][d]
  float* Ks   = (float*)(smem + 147456);   // 1 KB
  float* Zl   = (float*)(smem + 148480);   // 4 KB  [n][h]
  float* mg   = (float*)(smem + 152576);   // 128 KB f32
  bf*    msgT = f1t;
  bf*    lnT  = Qt;
  int tid = threadIdx.x, lane = tid & 31, wave = tid >> 5;
  int b = blockIdx.x >> 7;
  int n0 = (blockIdx.x & 127) * 128;
  size_t gb = ((size_t)b * NN + n0) * 256;
  int rb2 = (lane >> 4) << 3;
  for (int u = tid; u < 4096; u += 256)
    cp_async16((char*)f1t + u*16, (const char*)f1b + gb*2 + u*16);
  for (int i = tid; i < 8192; i += 256){
    int hh = i >> 10, vp = (i >> 5) & 31, d = i & 31;
    KVt[i] = f2bf(KVg[(((size_t)b * NH + hh) * 32 + d) * 32 + vp]);
  }
  Ks[tid] = Ksumg[b*256 + tid];
  wait_async();
  __syncthreads();
  // Q = elu(f1 @ wq^T) + 1
  for (int which = 0; which < 2; ++which){
    int nt = wave + which * 8;
    int col = nt*16 + (lane & 15);
    gemm_strip_k256(f1t, wq, nt, lane, [&](int mt, v8f acc){
      int rbase = mt*16 + rb2;
#pragma unroll
      for (int r = 0; r < 8; ++r){
        float q = acc[r]; q = q > 0.f ? q + 1.f : __expf(q);
        Qt[(rbase + r)*256 + col] = f2bf(q);
      }
    });
  }
  __syncthreads();
  // Z = 1 / (Q . Ksum + eps)
  for (int i = tid; i < 1024; i += 256){
    int n = i >> 3, hh = i & 7; float s = 0.f;
    for (int d = 0; d < 32; ++d) s += bf2f(Qt[n*256 + hh*32 + d]) * Ks[hh*32 + d];
    Zl[i] = 1.f / (s + 1e-6f);
  }
  __syncthreads();
  // msg = (Q_h @ KV_h) * Z * S   (overwrites f1t alias; f1t reads are done)
  for (int t = wave; t < 128; t += 8){
    int mt = t >> 4, nt = t & 15, hh = nt >> 1;
    v8f acc = {};
    acc = wmma_bf16(load_a(Qt + mt*16*256 + hh*32, 256, lane),
                    load_b(KVt + (hh*32 + (nt & 1)*16)*32, 32, lane), acc);
    int col = nt*16 + (lane & 15), rbase = mt*16 + rb2;
#pragma unroll
    for (int r = 0; r < 8; ++r){
      int row = rbase + r;
      msgT[row*256 + col] = f2bf(acc[r] * Zl[row*8 + hh] * 16384.f);
    }
  }
  __syncthreads();
  // merge GEMM
  for (int which = 0; which < 2; ++which){
    int nt = wave + which * 8;
    int col = nt*16 + (lane & 15);
    gemm_strip_k256(msgT, wmg, nt, lane, [&](int mt, v8f acc){
      int rbase = mt*16 + rb2;
#pragma unroll
      for (int r = 0; r < 8; ++r) mg[(rbase + r)*256 + col] = acc[r];
    });
  }
  __syncthreads();
  if (tid < 128){
    float s = 0.f, s2 = 0.f;
    for (int c = 0; c < 256; ++c){ float x = mg[tid*256 + c]; s += x; s2 += x*x; }
    float mu = s * (1.f/256.f), var = s2 * (1.f/256.f) - mu*mu;
    float inv = rsqrtf(var + 1e-5f);
    for (int c = 0; c < 256; ++c)
      lnT[tid*256 + c] = f2bf((mg[tid*256 + c] - mu) * inv * g1[c] + b1[c]);
  }
  __syncthreads();
  for (int u = tid; u < 4096; u += 256)
    st_async16((char*)ln1b + gb*2 + u*16, (const char*)lnT + u*16);
  wait_async();
}

// out = LN2( relu([f1, ln1] @ w1^T) @ w2^T ), stored [B,256,N]
__global__ void k_mlp(const bf* __restrict__ f1b, const bf* __restrict__ ln1b,
                      const bf* __restrict__ w1, const bf* __restrict__ w2,
                      const float* __restrict__ g2, const float* __restrict__ b2,
                      float* __restrict__ out){
  extern __shared__ char smem[];
  bf*    cat = (bf*)smem;                 // 128x512 (128 KB), aliased by `of`
  bf*    h1  = (bf*)(smem + 131072);      // 128x512 (128 KB), aliased by outT
  float* of  = (float*)smem;              // 128x256 f32
  float* outT = (float*)(smem + 131072);  // 256x128 f32 (alias h1)
  float* muA = (float*)(smem + 262144);   // 512 B
  float* invA = (float*)(smem + 262656);  // 512 B
  int tid = threadIdx.x, lane = tid & 31, wave = tid >> 5;
  int b = blockIdx.x >> 7;
  int n0 = (blockIdx.x & 127) * 128;
  size_t gb = ((size_t)b * NN + n0) * 256;
  for (int u = tid; u < 4096; u += 256){
    int row = u >> 5, cu = u & 31;
    cp_async16((char*)cat + row*1024 + cu*16,       (const char*)f1b  + gb*2 + row*512 + cu*16);
    cp_async16((char*)cat + row*1024 + 512 + cu*16, (const char*)ln1b + gb*2 + row*512 + cu*16);
  }
  wait_async();
  __syncthreads();
  for (int t = wave; t < 256; t += 8){
    int mt = t >> 5, nt = t & 31;
    v8f acc = {};
#pragma unroll
    for (int kk = 0; kk < 512; kk += 32)
      acc = wmma_bf16(load_a(cat + mt*16*512 + kk, 512, lane),
                      load_b(w1 + nt*16*512 + kk, 512, lane), acc);
    int col = nt*16 + (lane & 15), rbase = mt*16 + ((lane >> 4) << 3);
#pragma unroll
    for (int r = 0; r < 8; ++r){
      float x = acc[r];
      h1[(rbase + r)*512 + col] = f2bf(x > 0.f ? x : 0.f);
    }
  }
  __syncthreads();
  for (int t = wave; t < 128; t += 8){
    int mt = t >> 4, nt = t & 15;
    v8f acc = {};
#pragma unroll
    for (int kk = 0; kk < 512; kk += 32)
      acc = wmma_bf16(load_a(h1 + mt*16*512 + kk, 512, lane),
                      load_b(w2 + nt*16*512 + kk, 512, lane), acc);
    int col = nt*16 + (lane & 15), rbase = mt*16 + ((lane >> 4) << 3);
#pragma unroll
    for (int r = 0; r < 8; ++r) of[(rbase + r)*256 + col] = acc[r];
  }
  __syncthreads();
  if (tid < 128){
    float s = 0.f, s2 = 0.f;
    for (int c = 0; c < 256; ++c){ float x = of[tid*256 + c]; s += x; s2 += x*x; }
    float mu = s * (1.f/256.f), var = s2 * (1.f/256.f) - mu*mu;
    muA[tid] = mu; invA[tid] = rsqrtf(var + 1e-5f);
  }
  __syncthreads();
  {
    int c = tid; float gg = g2[c], bb = b2[c];
    for (int row = 0; row < 128; ++row)
      outT[c*128 + row] = (of[row*256 + c] - muA[row]) * invA[row] * gg + bb;
  }
  __syncthreads();
  for (int u = tid; u < 8192; u += 256){
    int c = u >> 5, w = u & 31;
    st_async16((char*)out + (((size_t)b*256 + c)*NN + n0)*4 + w*16,
               (const char*)outT + c*512 + w*16);
  }
  wait_async();
}

// ---------------------------------------------------------------------------
extern "C" void kernel_launch(void* const* d_in, const int* in_sizes, int n_in,
                              void* d_out, int out_size, void* d_ws, size_t ws_size,
                              hipStream_t stream){
  (void)in_sizes; (void)n_in; (void)out_size; (void)ws_size;
  const float* feat1 = (const float*)d_in[0];
  const float* feat2 = (const float*)d_in[2];
  const float* xyz2  = (const float*)d_in[3];
  const float* pw1   = (const float*)d_in[4];
  const float* pb1   = (const float*)d_in[5];
  const float* pw2   = (const float*)d_in[6];
  const float* pb2   = (const float*)d_in[7];
  const float* qw    = (const float*)d_in[8];
  const float* kw    = (const float*)d_in[9];
  const float* vw    = (const float*)d_in[10];
  const float* mw    = (const float*)d_in[11];
  const float* m1w   = (const float*)d_in[12];
  const float* m2w   = (const float*)d_in[13];
  const float* g1    = (const float*)d_in[14];
  const float* b1    = (const float*)d_in[15];
  const float* g2    = (const float*)d_in[16];
  const float* b2    = (const float*)d_in[17];
  float* out = (float*)d_out;
  char* ws = (char*)d_ws;

  size_t o = 0;
  bf* f1b   = (bf*)(ws + o); o += (size_t)BB * NN * 256 * 2;
  bf* f2b   = (bf*)(ws + o); o += (size_t)BB * SS * 256 * 2;
  bf* f2pb  = (bf*)(ws + o); o += (size_t)BB * SS * 256 * 2;
  bf* ln1b  = (bf*)(ws + o); o += (size_t)BB * NN * 256 * 2;
  bf* wqb   = (bf*)(ws + o); o += 256 * 256 * 2;
  bf* wkb   = (bf*)(ws + o); o += 256 * 256 * 2;
  bf* wvb   = (bf*)(ws + o); o += 256 * 256 * 2;
  bf* wmb   = (bf*)(ws + o); o += 256 * 256 * 2;
  bf* wp2b  = (bf*)(ws + o); o += 256 * 256 * 2;
  bf* w1b   = (bf*)(ws + o); o += 512 * 512 * 2;
  bf* w2b   = (bf*)(ws + o); o += 256 * 512 * 2;
  float* KVg   = (float*)(ws + o); o += (size_t)BB * NH * 32 * 32 * 4;
  float* Ksumg = (float*)(ws + o); o += (size_t)BB * 256 * 4;

  k_cvt<<<256, 256, 0, stream>>>(qw,  wqb, 65536);
  k_cvt<<<256, 256, 0, stream>>>(kw,  wkb, 65536);
  k_cvt<<<256, 256, 0, stream>>>(vw,  wvb, 65536);
  k_cvt<<<256, 256, 0, stream>>>(mw,  wmb, 65536);
  k_cvt<<<256, 256, 0, stream>>>(pw2, wp2b, 65536);
  k_cvt<<<1024, 256, 0, stream>>>(m1w, w1b, 262144);
  k_cvt<<<512, 256, 0, stream>>>(m2w, w2b, 131072);

  k_transpose<<<4096, 256, 0, stream>>>(feat1, f1b);
  k_transpose<<<4096, 256, 0, stream>>>(feat2, f2b);

  k_pos<<<BB * 128, 256, 198656, stream>>>(xyz2, pw1, pb1, wp2b, pb2, f2b, f2pb);

  (void)hipMemsetAsync(KVg, 0, (size_t)BB * NH * 32 * 32 * 4 + (size_t)BB * 256 * 4, stream);

  k_kv<<<BB * 16, 256, 262144, stream>>>(f2b, f2pb, wkb, wvb, KVg, Ksumg);

  k_attn<<<BB * 128, 256, 283648, stream>>>(f1b, wqb, wmb, KVg, Ksumg, g1, b1, ln1b);

  k_mlp<<<BB * 128, 256, 263168, stream>>>(f1b, ln1b, w1b, w2b, g2, b2, out);
}
